// MediumASTGCN_79671643341306
// MI455X (gfx1250) — compile-verified, matrix operand order checked by reading, not run on hardware
//
#include <hip/hip_runtime.h>
#include <hip/hip_bf16.h>

typedef _Float16 h16;
typedef __attribute__((ext_vector_type(16))) _Float16 v16h;
typedef __attribute__((ext_vector_type(8)))  _Float16 v8h;
typedef __attribute__((ext_vector_type(8)))  float    v8f;

constexpr int Bd = 4, Td = 12, Nd = 1024, Fin = 6, Ed = 16384, Cd = 64;
constexpr float EPSc = 1e-5f;

__device__ __forceinline__ float sigm(float x) { return 1.f / (1.f + __expf(-x)); }

__device__ __forceinline__ v16h frag_ld(const h16* p) {
  v8h lo = *(const v8h*)(p);
  v8h hi = *(const v8h*)(p + 16);
  return __builtin_shufflevector(lo, hi, 0, 1, 2, 3, 4, 5, 6, 7,
                                 8, 9, 10, 11, 12, 13, 14, 15);
}

// ---------------- f16 WMMA GEMM: C[m,n] = sum_k A[m,k]*B[n,k] (+epilogue) ------------
// Requirements (guaranteed by all call sites): K % 32 == 0, M % 32 == 0, N % 32 == 0,
// A rows (ldam) and B rows (ldbn) K-contiguous, 16-byte aligned.
// One wave32 computes a 32x32 output block as a 2x2 grid of 16x16 WMMA tiles:
// each A/B fragment is reused twice -> 4 v_wmma per 8 b128 loads per K-step.
// Fragment layout matches the CDNA5 16-bit operand lane swizzle
// (lanes 0-15: K 0-7/16-23; lanes 16-31: K 8-15/24-31).
__global__ __launch_bounds__(32) void k_wmma_gemm(
    const h16* __restrict__ A, const h16* __restrict__ Bm,
    const float* __restrict__ bias, float* __restrict__ C,
    int K, long aBatch, long bBatch, long cBatch,
    int ldam, int ldbn, int ldcm, int ldcn, int epi)
{
  int lane = threadIdx.x;
  int g = lane >> 4;          // K-half selector per CDNA5 16-bit operand layout
  int l = lane & 15;
  int tn = blockIdx.x * 32;
  int tm = blockIdx.y * 32;

  const h16* A0 = A + (long)blockIdx.z * aBatch + (long)(tm + l) * ldam + g * 8;
  const h16* A1 = A0 + (long)16 * ldam;
  const h16* B0 = Bm + (long)blockIdx.z * bBatch + (long)(tn + l) * ldbn + g * 8;
  const h16* B1 = B0 + (long)16 * ldbn;
  float* Cb = C + (long)blockIdx.z * cBatch;

  v8f acc00 = {0.f, 0.f, 0.f, 0.f, 0.f, 0.f, 0.f, 0.f};
  v8f acc01 = acc00, acc10 = acc00, acc11 = acc00;

  for (int k0 = 0; k0 < K; k0 += 32) {
    __builtin_prefetch(A0 + k0 + 32, 0, 1);   // speculative; dropped if OOB
    __builtin_prefetch(A1 + k0 + 32, 0, 1);
    __builtin_prefetch(B0 + k0 + 32, 0, 1);
    __builtin_prefetch(B1 + k0 + 32, 0, 1);
    v16h a0 = frag_ld(A0 + k0);
    v16h a1 = frag_ld(A1 + k0);
    v16h b0 = frag_ld(B0 + k0);
    v16h b1 = frag_ld(B1 + k0);
    acc00 = __builtin_amdgcn_wmma_f32_16x16x32_f16(false, a0, false, b0, (short)0, acc00, false, false);
    acc01 = __builtin_amdgcn_wmma_f32_16x16x32_f16(false, a0, false, b1, (short)0, acc01, false, false);
    acc10 = __builtin_amdgcn_wmma_f32_16x16x32_f16(false, a1, false, b0, (short)0, acc10, false, false);
    acc11 = __builtin_amdgcn_wmma_f32_16x16x32_f16(false, a1, false, b1, (short)0, acc11, false, false);
  }

  v8f accs[4] = {acc00, acc01, acc10, acc11};
#pragma unroll
  for (int ti = 0; ti < 2; ++ti)
#pragma unroll
    for (int tj = 0; tj < 2; ++tj) {
      v8f acc = accs[ti * 2 + tj];
#pragma unroll
      for (int r = 0; r < 8; ++r) {
        int m = tm + ti * 16 + r + g * 8;  // C/D layout: VGPR r -> M = r + 8*(lane/16)
        int n = tn + tj * 16 + l;          //             N = lane%16
        float v = acc[r];
        if (epi == 1) v += bias[m];
        else if (epi == 2) v = sigm(v + bias[m]);
        Cb[(long)m * ldcm + (long)n * ldcn] = v;
      }
    }
}

// ---------------- small / elementwise kernels ----------------------------------------
__global__ void k_perm(int n, const float* x, float* X) {
  int i = blockIdx.x * blockDim.x + threadIdx.x; if (i >= n) return;
  int t = i % Td; int f = (i / Td) % Fin; int nn = (i / (Td * Fin)) % Nd; int b = i / (Td * Fin * Nd);
  X[i] = x[((b * Td + t) * Nd + nn) * Fin + f];
}
__global__ void k_u1(int n, const float* X, const float* U1, int F, float* o) {
  int i = blockIdx.x * blockDim.x + threadIdx.x; if (i >= n) return;
  int f = i % F; int t = (i / F) % Td; int b = i / (F * Td);
  float s = 0.f;
  for (int nn = 0; nn < Nd; ++nn) s += X[((b * Nd + nn) * F + f) * Td + t] * U1[nn];
  o[i] = s;
}
__global__ void k_lhsT(int n, const float* l1, const float* U2, int F, float* o) {
  int i = blockIdx.x * blockDim.x + threadIdx.x; if (i >= n) return;
  int nn = i % Nd; int t = (i / Nd) % Td; int b = i / (Nd * Td);
  float s = 0.f;
  for (int f = 0; f < F; ++f) s += l1[(b * Td + t) * F + f] * U2[f * Nd + nn];
  o[i] = s;
}
__global__ void k_rhsT(int n, const float* X, const float* U3, int F, float* o) {
  int i = blockIdx.x * blockDim.x + threadIdx.x; if (i >= n) return;
  int t = i % Td; int nn = (i / Td) % Nd; int b = i / (Td * Nd);
  float s = 0.f;
  for (int f = 0; f < F; ++f) s += U3[f] * X[((b * Nd + nn) * F + f) * Td + t];
  o[i] = s;
}
__global__ void k_prodT(int n, const float* lT, const float* rT, float* o) {
  int i = blockIdx.x * blockDim.x + threadIdx.x; if (i >= n) return;
  int s_ = i % Td; int t = (i / Td) % Td; int b = i / (Td * Td);
  float s = 0.f;
  for (int nn = 0; nn < Nd; ++nn) s += lT[(b * Td + t) * Nd + nn] * rT[(b * Nd + nn) * Td + s_];
  o[i] = s;
}
__global__ void k_eatt(int n, const float* pT, const float* be, const float* Ve, float* E) {
  int i = blockIdx.x * blockDim.x + threadIdx.x; if (i >= n) return;
  int u = i % Td; int b = i / Td;
  float sg[Td], ev[Td];
  for (int s = 0; s < Td; ++s) sg[s] = sigm(pT[(b * Td + s) * Td + u] + be[s * Td + u]);
  float mx = -1e30f;
  for (int t = 0; t < Td; ++t) {
    float e = 0.f;
    for (int s = 0; s < Td; ++s) e += Ve[t * Td + s] * sg[s];
    ev[t] = e; mx = fmaxf(mx, e);
  }
  float sum = 0.f;
  for (int t = 0; t < Td; ++t) { ev[t] = __expf(ev[t] - mx); sum += ev[t]; }
  for (int t = 0; t < Td; ++t) E[(b * Td + t) * Td + u] = ev[t] / sum;
}
__global__ void k_xt(int n, const float* X, const float* E, int F, float* o) {
  int i = blockIdx.x * blockDim.x + threadIdx.x; if (i >= n) return;
  int s_ = i % Td; int f = (i / Td) % F; int nn = (i / (Td * F)) % Nd; int b = i / (Td * F * Nd);
  float s = 0.f;
  for (int t = 0; t < Td; ++t) s += X[((b * Nd + nn) * F + f) * Td + t] * E[(b * Td + t) * Td + s_];
  o[i] = s;
}
__global__ void k_slhs1(int n, const float* Xt, const float* W1, int F, float* o) {
  int i = blockIdx.x * blockDim.x + threadIdx.x; if (i >= n) return;
  int f = i % F; int nn = (i / F) % Nd; int b = i / (F * Nd);
  float s = 0.f;
  for (int t = 0; t < Td; ++t) s += Xt[((b * Nd + nn) * F + f) * Td + t] * W1[t];
  o[i] = s;
}
__global__ void k_slhs(int n, const float* l1, const float* W2, int F, float* o) {
  int i = blockIdx.x * blockDim.x + threadIdx.x; if (i >= n) return;
  int t = i % Td; int nn = (i / Td) % Nd; int b = i / (Td * Nd);
  float s = 0.f;
  for (int f = 0; f < F; ++f) s += l1[(b * Nd + nn) * F + f] * W2[f * Td + t];
  o[i] = s;
}
__global__ void k_srhs(int n, const float* Xt, const float* W3, int F, float* o) {
  int i = blockIdx.x * blockDim.x + threadIdx.x; if (i >= n) return;
  int t = i % Td; int nn = (i / Td) % Nd; int b = i / (Td * Nd);
  float s = 0.f;
  for (int f = 0; f < F; ++f) s += W3[f] * Xt[((b * Nd + nn) * F + f) * Td + t];
  o[i] = s;
}
__global__ void k_f2h(int n, const float* a, h16* o) {
  int i = blockIdx.x * blockDim.x + threadIdx.x; if (i >= n) return;
  o[i] = (h16)a[i];
}
// pack f32 rows of length inRow into f16 rows of length outRow (tail pre-zeroed)
__global__ void k_f2h_pad(int n, const float* a, int inRow, int outRow, h16* o) {
  int i = blockIdx.x * blockDim.x + threadIdx.x; if (i >= n) return;
  int r = i / inRow; int c = i % inRow;
  o[(long)r * outRow + c] = (h16)a[i];
}
// sigT[b,k,m] = sigmoid(prod[b,m,k] + bs[m,k])  (transposed store -> K-contiguous GEMM operand)
__global__ void k_sigbs_t(int n, const float* p, const float* bs, h16* o) {
  int i = blockIdx.x * blockDim.x + threadIdx.x; if (i >= n) return;
  int kc = i % Nd; int m = (i / Nd) % Nd; int b = i / (Nd * Nd);
  o[((long)b * Nd + kc) * Nd + m] = (h16)sigm(p[i] + bs[m * Nd + kc]);
}
__global__ void k_colsm(int n, const float* S, float* o) {
  int i = blockIdx.x * blockDim.x + threadIdx.x; if (i >= n) return;
  int c = i % Nd; int b = i / Nd; long base = (long)b * Nd * Nd + c;
  float mx = -1e30f;
  for (int nn = 0; nn < Nd; ++nn) mx = fmaxf(mx, S[base + (long)nn * Nd]);
  float sum = 0.f;
  for (int nn = 0; nn < Nd; ++nn) sum += __expf(S[base + (long)nn * Nd] - mx);
  float inv = 1.f / sum;
  for (int nn = 0; nn < Nd; ++nn) o[base + (long)nn * Nd] = __expf(S[base + (long)nn * Nd] - mx) * inv;
}
__global__ void k_deg(int n, const int* row, float* deg) {
  int i = blockIdx.x * blockDim.x + threadIdx.x; if (i >= n) return;
  atomicAdd(&deg[row[i]], 1.f);
}
__global__ void k_dis(int n, const float* deg, float* dis) {
  int i = blockIdx.x * blockDim.x + threadIdx.x; if (i >= n) return;
  float d = deg[i];
  dis[i] = (d > 0.f) ? rsqrtf(fmaxf(d, 1e-12f)) : 0.f;
}
__global__ void k_norm(int n, const int* row, const int* col, const float* dis,
                       const float* lamp, float* o) {
  int i = blockIdx.x * blockDim.x + threadIdx.x; if (i >= n) return;
  o[i] = (2.f / lamp[0]) * (-(dis[row[i]] * dis[col[i]]));
}
__global__ void k_attn(int n, const int* row, const int* col, const float* nm,
                       const float* S, float* o) {
  int i = blockIdx.x * blockDim.x + threadIdx.x; if (i >= n) return;
  int e = i % Ed; int b = i / Ed;
  o[i] = nm[e] * S[((long)b * Nd + row[e]) * Nd + col[e]];
}
__global__ void k_adiag(int n, const float* S, float* o) {
  int i = blockIdx.x * blockDim.x + threadIdx.x; if (i >= n) return;
  int nn = i % Nd; int b = i / Nd;
  o[i] = S[((long)b * Nd + nn) * Nd + nn];
}
__global__ void k_t0(int n, const float* X, const float* ad, int FT, float* o) {
  int i = blockIdx.x * blockDim.x + threadIdx.x; if (i >= n) return;
  int nn = (i / FT) % Nd; int b = i / (FT * Nd);
  o[i] = ad[b * Nd + nn] * X[i];
}
__global__ void k_scat(int n, const int* row, const int* col, const float* attn,
                       const float* z, int FT, float* agg) {
  int i = blockIdx.x * blockDim.x + threadIdx.x; if (i >= n) return;
  int ft = i % FT; int e = (i / FT) % Ed; int b = i / (FT * Ed);
  atomicAdd(&agg[((long)b * Nd + col[e]) * FT + ft],
            attn[b * Ed + e] * z[((long)b * Nd + row[e]) * FT + ft]);
}
__global__ void k_pfin(int n, const float* agg, const float* z, const float* Ts,
                       const float* ad, const float* lamp, float alpha, float beta,
                       int FT, float* o) {
  int i = blockIdx.x * blockDim.x + threadIdx.x; if (i >= n) return;
  int nn = (i / FT) % Nd; int b = i / (FT * Nd);
  float dcoef = 2.f / lamp[0] - 1.f;
  o[i] = alpha * (agg[i] + dcoef * ad[b * Nd + nn] * z[i]) + beta * Ts[i];
}
__global__ void k_cinit(int n, const float* cb, float* o) {
  int i = blockIdx.x * blockDim.x + threadIdx.x; if (i >= n) return;
  int oc = (i / Td) % Cd;
  o[i] = cb[oc];
}
__global__ void k_cacc(int n, const float* Tk, const float* cwk, int F, float* o) {
  int i = blockIdx.x * blockDim.x + threadIdx.x; if (i >= n) return;
  int t = i % Td; int oc = (i / Td) % Cd; int nn = (i / (Td * Cd)) % Nd; int b = i / (Td * Cd * Nd);
  long base = ((long)(b * Nd + nn) * F) * Td;
  float acc = o[i];
  for (int f = 0; f < F; ++f) acc += Tk[base + f * Td + t] * cwk[f * Cd + oc];
  o[i] = acc;
}
__global__ void k_relu(int n, float* a) {
  int i = blockIdx.x * blockDim.x + threadIdx.x; if (i >= n) return;
  a[i] = fmaxf(a[i], 0.f);
}
__global__ void k_rescon(int n, const float* Xin, const float* Xh, const float* rw,
                         const float* rb, const float* tw, const float* tb, int F, float* rc) {
  int i = blockIdx.x * blockDim.x + threadIdx.x; if (i >= n) return;
  int oc = i % Cd; int t = (i / Cd) % Td; int nn = (i / (Cd * Td)) % Nd; int b = i / (Cd * Td * Nd);
  float acc = rb[oc] + tb[oc];
  long xb = ((long)(b * Nd + nn) * F) * Td;
  for (int f = 0; f < F; ++f) acc += Xin[xb + f * Td + t] * rw[oc * F + f];
  long hb = ((long)(b * Nd + nn) * Cd) * Td;
  for (int c = 0; c < Cd; ++c) {
    int twb = (oc * Cd + c) * 3;
    if (t > 0)      acc += Xh[hb + c * Td + t - 1] * tw[twb + 0];
                    acc += Xh[hb + c * Td + t]     * tw[twb + 1];
    if (t < Td - 1) acc += Xh[hb + c * Td + t + 1] * tw[twb + 2];
  }
  rc[i] = fmaxf(acc, 0.f);
}
__global__ void k_ln(int n, const float* rc, const float* g, const float* be, float* Xo) {
  int i = blockIdx.x * blockDim.x + threadIdx.x; if (i >= n) return;
  int t = i % Td; int nn = (i / Td) % Nd; int b = i / (Td * Nd);
  long base = ((long)(b * Nd + nn) * Td + t) * Cd;
  float mu = 0.f;
  for (int o = 0; o < Cd; ++o) mu += rc[base + o];
  mu *= (1.f / Cd);
  float var = 0.f;
  for (int o = 0; o < Cd; ++o) { float d = rc[base + o] - mu; var += d * d; }
  var *= (1.f / Cd);
  float inv = rsqrtf(var + EPSc);
  for (int o = 0; o < Cd; ++o)
    Xo[((long)(b * Nd + nn) * Cd + o) * Td + t] = (rc[base + o] - mu) * inv * g[o] + be[o];
}
__global__ void k_fwr(int n, const float* fw, h16* o) {
  int i = blockIdx.x * blockDim.x + threadIdx.x; if (i >= n) return;
  int c = i % Cd; int t = (i / Cd) % Td; int p = i / (Cd * Td);
  o[(long)p * (Cd * Td) + c * Td + t] = (h16)fw[i];
}

// ---------------- host driver --------------------------------------------------------
#define EL(kern, n, ...) kern<<<dim3((unsigned)(((n) + 255) / 256)), dim3(256), 0, stream>>>((n), __VA_ARGS__)

extern "C" void kernel_launch(void* const* d_in, const int* in_sizes, int n_in,
                              void* d_out, int out_size, void* d_ws, size_t ws_size,
                              hipStream_t stream) {
  (void)in_sizes; (void)n_in; (void)out_size; (void)ws_size;
  const float* x   = (const float*)d_in[0];
  const int*   ei  = (const int*)d_in[1];
  const int*   rowp = ei;
  const int*   colp = ei + Ed;
  const float* fw   = (const float*)d_in[38];
  const float* fb   = (const float*)d_in[39];
  const float* fcw  = (const float*)d_in[40];
  const float* fcb  = (const float*)d_in[41];
  const float* lamp = (const float*)d_in[42];
  float* out = (float*)d_out;

  // workspace arena
  char* ws = (char*)d_ws;
  size_t off = 0;
  auto alloc = [&](size_t bytes) -> void* {
    off = (off + 255) & ~(size_t)255;
    void* p = ws + off;
    off += bytes;
    return p;
  };
  const size_t XSZ = (size_t)Bd * Nd * Cd * Td * 4;   // (B,N,64,T) f32
  const size_t NN  = (size_t)Bd * Nd * Nd;
  float* XA    = (float*)alloc(XSZ);
  float* XB    = (float*)alloc(XSZ);
  float* lhs1  = (float*)alloc((size_t)Bd * Td * Cd * 4);
  float* lhsT  = (float*)alloc((size_t)Bd * Td * Nd * 4);
  float* rhsT  = (float*)alloc((size_t)Bd * Nd * Td * 4);
  float* Eatt  = (float*)alloc((size_t)Bd * Td * Td * 4);
  float* prodT = (float*)alloc((size_t)Bd * Td * Td * 4);
  float* Xt    = (float*)alloc(XSZ);
  float* sl1   = (float*)alloc((size_t)Bd * Nd * Cd * 4);
  float* sl    = (float*)alloc((size_t)Bd * Nd * Td * 4);
  float* sr    = (float*)alloc((size_t)Bd * Nd * Td * 4);
  h16*   slhP  = (h16*)alloc((size_t)Bd * Nd * 32 * 2);   // K padded 12 -> 32
  h16*   srhP  = (h16*)alloc((size_t)Bd * Nd * 32 * 2);
  float* prodB = (float*)alloc(NN * 4);          // prod, then S_raw (aliased)
  h16*   sigT  = (h16*)alloc(NN * 2);            // transposed sigmoid; outT_h in final phase
  h16*   Vsh   = (h16*)alloc((size_t)Nd * Nd * 2); // also fc_w f16 in final phase
  float* Satt  = (float*)alloc(NN * 4);          // also out_buf in final phase
  float* deg   = (float*)alloc((size_t)Nd * 4);
  float* dis   = (float*)alloc((size_t)Nd * 4);
  float* nrm   = (float*)alloc((size_t)Ed * 4);
  float* attn  = (float*)alloc((size_t)Bd * Ed * 4);
  float* ad    = (float*)alloc((size_t)Bd * Nd * 4);
  float* T0b   = (float*)alloc(XSZ);             // also fwr_h in final phase
  float* T2b   = (float*)alloc(XSZ);             // also X2h in final phase
  float* aggb  = (float*)alloc(XSZ);             // also rc buffer (aliased)
  float* Xh    = (float*)alloc(XSZ);
  float* T1b   = Xt;                             // Xt dead by cheb time
  float* rc    = aggb;                           // agg dead after T2
  h16*   fwrh  = (h16*)T0b;                      // final phase reuse
  h16*   X2h   = (h16*)T2b;                      // final phase reuse
  float* outbf = Satt;
  h16*   outTh = sigT;
  h16*   fcwh  = Vsh;

  auto gemm = [&](const h16* A, const h16* Bm, const float* bias, float* Cc,
                  int M, int Nn, int K, long aB, long bB, long cB,
                  int ldam, int ldbn, int ldcm, int ldcn, int epi) {
    dim3 g((unsigned)(Nn / 32), (unsigned)(M / 32), Bd);
    k_wmma_gemm<<<g, dim3(32), 0, stream>>>(A, Bm, bias, Cc, K, aB, bB, cB,
                                            ldam, ldbn, ldcm, ldcn, epi);
  };

  // zero the K-padding of the prod GEMM operands once (tails stay zero both blocks)
  hipMemsetAsync(slhP, 0, (size_t)Bd * Nd * 32 * 2, stream);
  hipMemsetAsync(srhP, 0, (size_t)Bd * Nd * 32 * 2, stream);

  // degree / Laplacian normalization (shared by both blocks)
  hipMemsetAsync(deg, 0, (size_t)Nd * 4, stream);
  EL(k_deg, Ed, rowp, deg);
  EL(k_dis, Nd, deg, dis);
  EL(k_norm, Ed, rowp, colp, dis, lamp, nrm);

  auto blockf = [&](int F, int pb, const float* Xin, float* Xout) {
    const float* U1 = (const float*)d_in[pb + 0];
    const float* U2 = (const float*)d_in[pb + 1];
    const float* U3 = (const float*)d_in[pb + 2];
    const float* be = (const float*)d_in[pb + 3];
    const float* Ve = (const float*)d_in[pb + 4];
    const float* W1 = (const float*)d_in[pb + 5];
    const float* W2 = (const float*)d_in[pb + 6];
    const float* W3 = (const float*)d_in[pb + 7];
    const float* bs = (const float*)d_in[pb + 8];
    const float* Vs = (const float*)d_in[pb + 9];
    const float* cw = (const float*)d_in[pb + 10];
    const float* cb = (const float*)d_in[pb + 11];
    const float* tw = (const float*)d_in[pb + 12];
    const float* tb = (const float*)d_in[pb + 13];
    const float* rw = (const float*)d_in[pb + 14];
    const float* rb = (const float*)d_in[pb + 15];
    const float* lg = (const float*)d_in[pb + 16];
    const float* lb = (const float*)d_in[pb + 17];
    const int FT = F * Td;

    // temporal attention
    EL(k_u1, Bd * Td * F, Xin, U1, F, lhs1);
    EL(k_lhsT, Bd * Td * Nd, lhs1, U2, F, lhsT);
    EL(k_rhsT, Bd * Nd * Td, Xin, U3, F, rhsT);
    EL(k_prodT, Bd * Td * Td, lhsT, rhsT, prodT);
    EL(k_eatt, Bd * Td, prodT, be, Ve, Eatt);
    EL(k_xt, Bd * Nd * F * Td, Xin, Eatt, F, Xt);

    // spatial attention
    EL(k_slhs1, Bd * Nd * F, Xt, W1, F, sl1);
    EL(k_slhs, Bd * Nd * Td, sl1, W2, F, sl);
    EL(k_srhs, Bd * Nd * Td, Xt, W3, F, sr);
    EL(k_f2h_pad, Bd * Nd * Td, sl, Td, 32, slhP);
    EL(k_f2h_pad, Bd * Nd * Td, sr, Td, 32, srhP);
    // prod[b,n,m] = sum_t lhs[b,n,t]*rhs[b,m,t]  (WMMA, K padded to 32)
    gemm(slhP, srhP, nullptr, prodB, Nd, Nd, 32,
         (long)Nd * 32, (long)Nd * 32, (long)Nd * Nd,
         32, 32, Nd, 1, 0);
    EL(k_sigbs_t, (int)NN, prodB, bs, sigT);
    EL(k_f2h, Nd * Nd, Vs, Vsh);
    // S_raw[b,n,k] = sum_m Vs[n,m]*sigT[b,k,m]   (WMMA, K=1024) -> prodB reused
    gemm(Vsh, sigT, nullptr, prodB, Nd, Nd, Nd,
         0, (long)Nd * Nd, (long)Nd * Nd,
         Nd, Nd, Nd, 1, 0);
    EL(k_colsm, Bd * Nd, prodB, Satt);

    // Chebyshev graph conv with attention
    EL(k_attn, Bd * Ed, rowp, colp, nrm, Satt, attn);
    EL(k_adiag, Bd * Nd, Satt, ad);
    EL(k_t0, Bd * Nd * FT, Xin, ad, FT, T0b);
    hipMemsetAsync(aggb, 0, (size_t)Bd * Nd * FT * 4, stream);
    EL(k_scat, Bd * Ed * FT, rowp, colp, attn, T0b, FT, aggb);
    EL(k_pfin, Bd * Nd * FT, aggb, T0b, T0b, ad, lamp, 1.f, 0.f, FT, T1b);
    hipMemsetAsync(aggb, 0, (size_t)Bd * Nd * FT * 4, stream);
    EL(k_scat, Bd * Ed * FT, rowp, colp, attn, T1b, FT, aggb);
    EL(k_pfin, Bd * Nd * FT, aggb, T1b, T0b, ad, lamp, 2.f, -1.f, FT, T2b);
    EL(k_cinit, Bd * Nd * Cd * Td, cb, Xh);
    EL(k_cacc, Bd * Nd * Cd * Td, T0b, cw + 0 * F * Cd, F, Xh);
    EL(k_cacc, Bd * Nd * Cd * Td, T1b, cw + 1 * F * Cd, F, Xh);
    EL(k_cacc, Bd * Nd * Cd * Td, T2b, cw + 2 * F * Cd, F, Xh);
    EL(k_relu, Bd * Nd * Cd * Td, Xh);

    // temporal conv + residual + relu + layer norm
    EL(k_rescon, Bd * Nd * Td * Cd, Xin, Xh, rw, rb, tw, tb, F, rc);
    EL(k_ln, Bd * Nd * Td, rc, lg, lb, Xout);
  };

  EL(k_perm, Bd * Nd * Fin * Td, x, XA);
  blockf(Fin, 2, XA, XB);
  blockf(Cd, 20, XB, XA);

  // final projection: out[b,p,n] = sum_{t,c} X[b,n,c,t]*fw[p,t,c] + fb[p]
  EL(k_fwr, Nd * Td * Cd, fw, fwrh);
  EL(k_f2h, Bd * Nd * Cd * Td, XA, X2h);
  gemm(fwrh, X2h, fb, outbf, Nd, Nd, Cd * Td,
       0, (long)Nd * Cd * Td, (long)Nd * Nd,
       Cd * Td, Cd * Td, 1, Nd, 1);               // store transposed: [b, n, p]

  // fc head: out2[b,n,j] = sigmoid(sum_p outT[b,n,p]*fc_w[j,p] + fc_b[j])
  EL(k_f2h, (int)NN, outbf, outTh);
  EL(k_f2h, Nd * Nd, fcw, fcwh);
  gemm(fcwh, outTh, fcb, out, Nd, Nd, Nd,
       0, (long)Nd * Nd, (long)Nd * Nd,
       Nd, Nd, 1, Nd, 2);                         // store [b, n, j], sigmoid epilogue
}
